// ARGRModule_85727547228408
// MI455X (gfx1250) — matrix operand order repaired and checked by reading, not compiled, not run on hardware
//
#include <hip/hip_runtime.h>

// ---------------------------------------------------------------------------
// CDNA5 / gfx1250 implementation.
// Heavy math: v_wmma_f32_16x16x32_bf16 (f32 accumulate) fed from LDS panels
// staged with GLOBAL_LOAD_ASYNC_TO_LDS_B128 (ASYNCcnt, double buffered).
// Problem constants: B=2, C=256, H=W=128, Hd=Wd=64, N=4096 nodes/sample.
// Requires ~105 MB of workspace.
// ---------------------------------------------------------------------------

typedef __attribute__((ext_vector_type(16))) __bf16 v16bf;
typedef __attribute__((ext_vector_type(8)))  __bf16 v8bf;
typedef __attribute__((ext_vector_type(8)))  float  v8f;
typedef int v4i_ __attribute__((ext_vector_type(4)));

#define Bsz   2
#define Cdim  256
#define Nnode 4096
#define HD    64
#define LEAKY 0.2f

// LDS panel geometry: rows padded to 80B (20 banks) so 16-row b128 column
// reads are bank-conflict-free; 64B K-window per row in 4 x 16B chunks.
#define LDS_ROW   80
#define A_PANEL   (128 * LDS_ROW)            // 10240 B
#define B_PANEL   (64 * LDS_ROW)             // 5120 B
#define BUF_SZ    (A_PANEL + B_PANEL)        // 15360 B
// total LDS: 2 * 15360 = 30720 B per workgroup

#if __has_builtin(__builtin_amdgcn_global_load_async_to_lds_b128)
#define ASYNC_CPY16(g, l)                                                     \
    __builtin_amdgcn_global_load_async_to_lds_b128(                           \
        (__attribute__((address_space(1))) v4i_*)(g),                         \
        (__attribute__((address_space(3))) v4i_*)(l), 0, 0)
#else
#define ASYNC_CPY16(g, l)                                                     \
    asm volatile("global_load_async_to_lds_b128 %0, %1, off"                  \
                 :: "v"((unsigned)(unsigned long)(l)), "v"(g) : "memory")
#endif

#if __has_builtin(__builtin_amdgcn_s_wait_asynccnt)
#define WAIT_ASYNC0() __builtin_amdgcn_s_wait_asynccnt(0)
#else
#define WAIT_ASYNC0() asm volatile("s_wait_asynccnt 0x0" ::: "memory")
#endif

#define CAT16(lo, hi) __builtin_shufflevector(lo, hi,                         \
    0,1,2,3,4,5,6,7,8,9,10,11,12,13,14,15)

// ---------------------------------------------------------------------------
// Generic bf16 GEMM:  C[i][j] = sum_k A[i][k] * W[j][k]   (C = A @ W^T)
// A,W row-major bf16 (lda/ldw in elements). Output element (i,j) stored at
// row*out_rs + col*out_cs into outF (f32) if non-null else outB (bf16).
// Grid (Ncols/64, M/128), block 256 = 8 waves; wave tile 16 rows x 64 cols.
// Per K-step (32): stage A panel (128x32) + B panel (64x32) into LDS via
// async-to-LDS copies (3 x b128 per thread), double buffered.  All LDS reads
// of a step are issued before the 4 WMMAs so ds latency overlaps WMMA issue.
// No lane divergence -> EXEC all ones at WMMA.
// ---------------------------------------------------------------------------
__global__ __launch_bounds__(256) void gemm_bf16_wmma(
    const __bf16* __restrict__ A, int lda,
    const __bf16* __restrict__ W, int ldw,
    float* __restrict__ outF, __bf16* __restrict__ outB,
    long out_rs, long out_cs, int K)
{
    __shared__ char smem[2 * BUF_SZ];

    const int tid  = threadIdx.x;
    const int lane = tid & 31;
    const int wave = tid >> 5;
    const int i0   = blockIdx.y * 128 + wave * 16;
    const int j0   = blockIdx.x * 64;
    const int ksel = lane >> 4;       // 0: lanes 0-15, 1: lanes 16-31
    const int mrow = lane & 15;

    const __bf16* Ablk = A + (long)blockIdx.y * 128 * lda;   // 128-row panel
    const __bf16* Wblk = W + (long)j0 * ldw;                 // 64-row panel

    // per-thread async copy coordinates (fixed across K steps)
    const int brow = tid >> 2, bch = tid & 3;                // B: 64x4 chunks
    const int ar0 = tid >> 1, ac0 = (tid & 1) * 2;           // A: 128 rows x 4 chunks,
                                                             //    2 consecutive chunks/thread
    v8f c0 = {}, c1 = {}, c2 = {}, c3 = {};

    const int S = K >> 5;

    // prologue: stage K-step 0 into buffer 0
    {
        char* bufA = smem;
        char* bufB = smem + A_PANEL;
        ASYNC_CPY16(Wblk + (long)brow * ldw + bch * 8, bufB + brow * LDS_ROW + bch * 16);
        ASYNC_CPY16(Ablk + (long)ar0 * lda + ac0 * 8,       bufA + ar0 * LDS_ROW + ac0 * 16);
        ASYNC_CPY16(Ablk + (long)ar0 * lda + (ac0 + 1) * 8, bufA + ar0 * LDS_ROW + (ac0 + 1) * 16);
    }

    for (int s = 0; s < S; ++s) {
        WAIT_ASYNC0();          // own group-s copies landed
        __syncthreads();        // everyone's group-s data visible; buf[(s+1)&1] free
        const char* bufA = smem + (s & 1) * BUF_SZ;
        const char* bufB = bufA + A_PANEL;

        if (s + 1 < S) {        // prefetch group s+1 into the other buffer
            const int k = (s + 1) << 5;
            char* nA = smem + ((s + 1) & 1) * BUF_SZ;
            char* nB = nA + A_PANEL;
            ASYNC_CPY16(Wblk + (long)brow * ldw + k + bch * 8, nB + brow * LDS_ROW + bch * 16);
            ASYNC_CPY16(Ablk + (long)ar0 * lda + k + ac0 * 8,       nA + ar0 * LDS_ROW + ac0 * 16);
            ASYNC_CPY16(Ablk + (long)ar0 * lda + k + (ac0 + 1) * 8, nA + ar0 * LDS_ROW + (ac0 + 1) * 16);
        }

        // ---- issue ALL LDS reads of this step first (one clause), then WMMA.
        // A lane layout (16-bit A 16x32): chunks {kb..kb+7},{kb+16..kb+23},
        // kb = ksel*8 -> LDS bytes ksel*16 and ksel*16+32 of the 64B row.
        const char* arow = bufA + (wave * 16 + mrow) * LDS_ROW + ksel * 16;
        v8bf alo = *(const v8bf*)(arow);
        v8bf ahi = *(const v8bf*)(arow + 32);
        // B lane layout (16-bit B 32x16): same K chunks as A, of W row
        // (j0 + t*16 + mrow), bytes ksel*16 and ksel*16+32.
        const char* brow0 = bufB + mrow * LDS_ROW + ksel * 16;
        v8bf b0lo = *(const v8bf*)(brow0);
        v8bf b0hi = *(const v8bf*)(brow0 + 32);
        v8bf b1lo = *(const v8bf*)(brow0 + 16 * LDS_ROW);
        v8bf b1hi = *(const v8bf*)(brow0 + 16 * LDS_ROW + 32);
        v8bf b2lo = *(const v8bf*)(brow0 + 32 * LDS_ROW);
        v8bf b2hi = *(const v8bf*)(brow0 + 32 * LDS_ROW + 32);
        v8bf b3lo = *(const v8bf*)(brow0 + 48 * LDS_ROW);
        v8bf b3hi = *(const v8bf*)(brow0 + 48 * LDS_ROW + 32);

        v16bf a  = CAT16(alo, ahi);
        v16bf b0 = CAT16(b0lo, b0hi);
        v16bf b1 = CAT16(b1lo, b1hi);
        v16bf b2 = CAT16(b2lo, b2hi);
        v16bf b3 = CAT16(b3lo, b3hi);

        c0 = __builtin_amdgcn_wmma_f32_16x16x32_bf16(false, a, false, b0, (short)0, c0, false, false);
        c1 = __builtin_amdgcn_wmma_f32_16x16x32_bf16(false, a, false, b1, (short)0, c1, false, false);
        c2 = __builtin_amdgcn_wmma_f32_16x16x32_bf16(false, a, false, b2, (short)0, c2, false, false);
        c3 = __builtin_amdgcn_wmma_f32_16x16x32_bf16(false, a, false, b3, (short)0, c3, false, false);
    }

    // C/D layout: VGPR r -> M=r (lanes 0-15) / M=8+r (lanes 16-31), N=lane&15.
    const int n  = lane & 15;
    const int mo = (lane >> 4) * 8;
    const long rowbase = (long)(i0 + mo) * out_rs;
    for (int r = 0; r < 8; ++r) {
        const long rb = rowbase + (long)r * out_rs;
        if (outF) {
            outF[rb + (long)(j0 +      n) * out_cs] = c0[r];
            outF[rb + (long)(j0 + 16 + n) * out_cs] = c1[r];
            outF[rb + (long)(j0 + 32 + n) * out_cs] = c2[r];
            outF[rb + (long)(j0 + 48 + n) * out_cs] = c3[r];
        } else {
            outB[rb + (long)(j0 +      n) * out_cs] = (__bf16)c0[r];
            outB[rb + (long)(j0 + 16 + n) * out_cs] = (__bf16)c1[r];
            outB[rb + (long)(j0 + 32 + n) * out_cs] = (__bf16)c2[r];
            outB[rb + (long)(j0 + 48 + n) * out_cs] = (__bf16)c3[r];
        }
    }
}

// 2x2 maxpool: feat (B,C,128,128) -> X (B,N,C) in f32 and bf16 (channel-last).
__global__ __launch_bounds__(256) void pool_kernel(
    const float* __restrict__ fm, float* __restrict__ Xf, __bf16* __restrict__ Xb)
{
    const int idx = blockIdx.x * 256 + threadIdx.x;     // (b,n,c) flat
    const int c = idx & 255;
    const int n = (idx >> 8) & (Nnode - 1);
    const int b = idx >> 20;
    const int y = n >> 6, x = n & 63;
    const float* p = fm + (((long)(b * Cdim + c) * 128) + 2 * y) * 128 + 2 * x;
    float m = fmaxf(fmaxf(p[0], p[1]), fmaxf(p[128], p[129]));
    Xf[idx] = m;
    Xb[idx] = (__bf16)m;
}

// cos(phi), sin(phi) per node without atan2.
__global__ __launch_bounds__(256) void phi_kernel(float* __restrict__ cph, float* __restrict__ sph)
{
    const int n = blockIdx.x * 256 + threadIdx.x;
    const float dy = (float)(n >> 6) - 31.5f;
    const float dx = (float)(n & 63) - 31.5f;
    const float r = sqrtf(dx * dx + dy * dy);
    cph[n] = dx / r;
    sph[n] = dy / r;
}

__global__ __launch_bounds__(256) void convert_kernel(
    const float* __restrict__ src, __bf16* __restrict__ dst)
{
    const int i = blockIdx.x * 256 + threadIdx.x;
    dst[i] = (__bf16)src[i];
}

// u[i] = X[i]·w_u, v[i] = X[i]·w_v   (both samples; idx = b*N + i)
__global__ __launch_bounds__(256) void uv_kernel(
    const float* __restrict__ Xf, const float* __restrict__ Wg,
    float* __restrict__ u, float* __restrict__ v)
{
    const int idx = blockIdx.x * 256 + threadIdx.x;
    const float* xr = Xf + (long)idx * Cdim;
    float su = 0.f, sv = 0.f;
    for (int c = 0; c < Cdim; ++c) {
        su += xr[c] * Wg[c];
        sv += xr[c] * Wg[Cdim + c];
    }
    u[idx] = su;
    v[idx] = sv;
}

// Row softmax over 4096-wide f32 rows (a block of rows of e_g) -> bf16 A.
__global__ __launch_bounds__(256) void row_softmax(
    const float* __restrict__ Ein, __bf16* __restrict__ Aout, int rowOffset)
{
    __shared__ float red[256];
    const int row = blockIdx.x;
    const int tid = threadIdx.x;
    const float* e = Ein + (long)row * Nnode;
    float vals[16];
    float m = -3.4e38f;
    for (int t = 0; t < 16; ++t) { vals[t] = e[tid + t * 256]; m = fmaxf(m, vals[t]); }
    red[tid] = m; __syncthreads();
    for (int s = 128; s > 0; s >>= 1) { if (tid < s) red[tid] = fmaxf(red[tid], red[tid + s]); __syncthreads(); }
    const float M = red[0]; __syncthreads();
    float sum = 0.f;
    for (int t = 0; t < 16; ++t) { vals[t] = __expf(vals[t] - M); sum += vals[t]; }
    red[tid] = sum; __syncthreads();
    for (int s = 128; s > 0; s >>= 1) { if (tid < s) red[tid] += red[tid + s]; __syncthreads(); }
    const float inv = 1.0f / red[0];
    __bf16* out = Aout + (long)(rowOffset + row) * Nnode;
    for (int t = 0; t < 16; ++t) out[tid + t * 256] = (__bf16)(vals[t] * inv);
}

// e_l column-softmax stats (axis=0): one block per column j, recompute e.
__global__ __launch_bounds__(256) void el_stats(
    const float* __restrict__ u, const float* __restrict__ v,
    const float* __restrict__ cph, const float* __restrict__ sph,
    const float* __restrict__ Wg, float* __restrict__ colmax, float* __restrict__ colsum)
{
    __shared__ float red[256];
    const int j = blockIdx.x;
    const int tid = threadIdx.x;
    const float wc = Wg[2 * Cdim], ws = Wg[2 * Cdim + 1];
    const float vj = v[j], cj = cph[j], sj = sph[j];
    float m = -3.4e38f;
    for (int i = tid; i < Nnode; i += 256) {
        float e = u[i] + vj + wc * (cph[i] * cj + sph[i] * sj)
                          + ws * (sph[i] * cj - cph[i] * sj);
        e = (e >= 0.f) ? e : LEAKY * e;
        m = fmaxf(m, e);
    }
    red[tid] = m; __syncthreads();
    for (int s = 128; s > 0; s >>= 1) { if (tid < s) red[tid] = fmaxf(red[tid], red[tid + s]); __syncthreads(); }
    const float M = red[0]; __syncthreads();
    float sum = 0.f;
    for (int i = tid; i < Nnode; i += 256) {
        float e = u[i] + vj + wc * (cph[i] * cj + sph[i] * sj)
                          + ws * (sph[i] * cj - cph[i] * sj);
        e = (e >= 0.f) ? e : LEAKY * e;
        sum += __expf(e - M);
    }
    red[tid] = sum; __syncthreads();
    for (int s = 128; s > 0; s >>= 1) { if (tid < s) red[tid] += red[tid + s]; __syncthreads(); }
    if (tid == 0) { colmax[j] = M; colsum[j] = red[0]; }
}

// Fill A_l (bf16, row-major [i][j]) from recomputed e_l + column stats.
__global__ __launch_bounds__(256) void el_fill(
    const float* __restrict__ u, const float* __restrict__ v,
    const float* __restrict__ cph, const float* __restrict__ sph,
    const float* __restrict__ Wg, const float* __restrict__ colmax,
    const float* __restrict__ colsum, __bf16* __restrict__ Aout)
{
    const int j = blockIdx.x * 256 + threadIdx.x;
    const int i = blockIdx.y;
    const float wc = Wg[2 * Cdim], ws = Wg[2 * Cdim + 1];
    const float ci = cph[i], si = sph[i];
    float e = u[i] + v[j] + wc * (ci * cph[j] + si * sph[j])
                      + ws * (si * cph[j] - ci * sph[j]);
    e = (e >= 0.f) ? e : LEAKY * e;
    Aout[(long)i * Nnode + j] = (__bf16)(__expf(e - colmax[j]) / colsum[j]);
}

// per-head ReLU + sum over 3 heads: C768[i][j + 256k] -> Fb[i][colOff + j]
__global__ __launch_bounds__(256) void relu_heads(
    const float* __restrict__ C768, __bf16* __restrict__ Fb, int colOff)
{
    const int idx = blockIdx.x * 256 + threadIdx.x;   // i*256 + c
    const int i = idx >> 8, c = idx & 255;
    const float* p = C768 + (long)i * 768 + c;
    float s = fmaxf(p[0], 0.f) + fmaxf(p[256], 0.f) + fmaxf(p[512], 0.f);
    Fb[(long)i * 768 + colOff + c] = (__bf16)s;
}

// Copy X (bf16) into fused buffer columns 0..255 for sample b.
__global__ __launch_bounds__(256) void fb_init(
    const __bf16* __restrict__ Xb, __bf16* __restrict__ Fb, int b)
{
    const int idx = blockIdx.x * 256 + threadIdx.x;   // i*256 + c
    const int i = idx >> 8, c = idx & 255;
    Fb[(long)i * 768 + c] = Xb[((long)b * Nnode + i) * Cdim + c];
}

// bias + 2x2 upsample (kern == 0.25 everywhere) into final output (b,c,128,128)
__global__ __launch_bounds__(256) void upsample_kernel(
    const float* __restrict__ down, const float* __restrict__ bias,
    float* __restrict__ out, int b)
{
    const int idx = blockIdx.x * 256 + threadIdx.x;   // c*128*128 + yy*128 + xx
    const int xx = idx & 127;
    const int yy = (idx >> 7) & 127;
    const int c  = idx >> 14;
    const int n  = (yy >> 1) * HD + (xx >> 1);
    const float val = 0.25f * (down[(long)n * Cdim + c] + bias[c]);
    out[(((long)(b * Cdim + c)) * 128 + yy) * 128 + xx] = val;
}

// ---------------------------------------------------------------------------
static inline void launch_gemm(const __bf16* A, int lda, const __bf16* W, int ldw,
                               float* outF, __bf16* outB, long rs, long cs,
                               int M, int Ncols, int K, hipStream_t s)
{
    dim3 grid(Ncols / 64, M / 128);
    gemm_bf16_wmma<<<grid, 256, 0, s>>>(A, lda, W, ldw, outF, outB, rs, cs, K);
}

extern "C" void kernel_launch(void* const* d_in, const int* in_sizes, int n_in,
                              void* d_out, int out_size, void* d_ws, size_t ws_size,
                              hipStream_t stream)
{
    (void)in_sizes; (void)n_in; (void)out_size; (void)ws_size;
    const float* feat   = (const float*)d_in[0];
    const float* Wtheta = (const float*)d_in[1];
    const float* Wphi   = (const float*)d_in[2];
    const float* gcb    = (const float*)d_in[3];
    const float* Wgamma = (const float*)d_in[4];
    const float* ltb    = (const float*)d_in[5];
    const float* fusew  = (const float*)d_in[6];
    const float* fuseb  = (const float*)d_in[7];
    float* out = (float*)d_out;

    char* p = (char*)d_ws;
    auto alloc = [&](size_t bytes) { void* r = (void*)p; p += (bytes + 255) & ~(size_t)255; return r; };

    float*  Xf     = (float*) alloc((size_t)Bsz * Nnode * Cdim * 4);
    __bf16* Xb     = (__bf16*)alloc((size_t)Bsz * Nnode * Cdim * 2);
    float*  cph    = (float*) alloc(Nnode * 4);
    float*  sph    = (float*) alloc(Nnode * 4);
    __bf16* Wb     = (__bf16*)alloc((size_t)2048 * Cdim * 2);      // theta,phi,g0..2,l0..2 rows
    __bf16* Wfuse  = (__bf16*)alloc((size_t)Cdim * 768 * 2);
    float*  u      = (float*) alloc((size_t)Bsz * Nnode * 4);
    float*  v      = (float*) alloc((size_t)Bsz * Nnode * 4);
    __bf16* TP     = (__bf16*)alloc((size_t)Nnode * 512 * 2);      // per-sample: [T|P]
    __bf16* Yt     = (__bf16*)alloc((size_t)1536 * Nnode * 2);     // per-sample: Y transposed
    float*  Af     = (float*) alloc((size_t)1024 * Nnode * 4);     // e_g row-block
    __bf16* Abf    = (__bf16*)alloc((size_t)Nnode * Nnode * 2);    // A_g / A_l bf16
    float*  C768   = (float*) alloc((size_t)Nnode * 768 * 4);
    __bf16* Fb     = (__bf16*)alloc((size_t)Nnode * 768 * 2);      // fused [X|g|l]
    float*  down   = (float*) alloc((size_t)Nnode * Cdim * 4);
    float*  colmax = (float*) alloc(Nnode * 4);
    float*  colsum = (float*) alloc(Nnode * 4);

    // ---- global preprocessing ------------------------------------------------
    pool_kernel<<<(Bsz * Nnode * Cdim) / 256, 256, 0, stream>>>(feat, Xf, Xb);
    phi_kernel<<<Nnode / 256, 256, 0, stream>>>(cph, sph);
    convert_kernel<<<(Cdim * Cdim) / 256,      256, 0, stream>>>(Wtheta, Wb);
    convert_kernel<<<(Cdim * Cdim) / 256,      256, 0, stream>>>(Wphi,   Wb + 1 * Cdim * Cdim);
    convert_kernel<<<(3 * Cdim * Cdim) / 256,  256, 0, stream>>>(gcb,    Wb + 2 * Cdim * Cdim);
    convert_kernel<<<(3 * Cdim * Cdim) / 256,  256, 0, stream>>>(ltb,    Wb + 5 * Cdim * Cdim);
    convert_kernel<<<(Cdim * 768) / 256,       256, 0, stream>>>(fusew,  Wfuse);
    uv_kernel<<<(Bsz * Nnode) / 256, 256, 0, stream>>>(Xf, Wgamma, u, v);

    // ---- per-sample pipeline -------------------------------------------------
    for (int b = 0; b < Bsz; ++b) {
        const __bf16* Xs = Xb + (size_t)b * Nnode * Cdim;
        const float*  us = u + (size_t)b * Nnode;
        const float*  vs = v + (size_t)b * Nnode;

        // T|P = X @ [Wtheta;Wphi]^T   (bf16 out, row-major 4096x512)
        launch_gemm(Xs, Cdim, Wb, Cdim, nullptr, TP, 512, 1, Nnode, 512, Cdim, stream);
        // Yt[j][k] = (X @ heads^T)[k][j]  (transposed bf16 store, 1536x4096)
        launch_gemm(Xs, Cdim, Wb + 512 * Cdim, Cdim, nullptr, Yt, 1, Nnode, Nnode, 1536, Cdim, stream);

        // e_g = T @ P^T in 1024-row blocks; row softmax -> A_g (bf16)
        for (int rb = 0; rb < 4; ++rb) {
            launch_gemm(TP + (size_t)rb * 1024 * 512, 512, TP + 256, 512,
                        Af, nullptr, Nnode, 1, 1024, Nnode, Cdim, stream);
            row_softmax<<<1024, 256, 0, stream>>>(Af, Abf, rb * 1024);
        }
        // g_all = A_g @ Y (768 cols), then per-head ReLU-sum -> Fb[:,256:512]
        launch_gemm(Abf, Nnode, Yt, Nnode, C768, nullptr, 768, 1, Nnode, 768, Nnode, stream);
        relu_heads<<<(Nnode * Cdim) / 256, 256, 0, stream>>>(C768, Fb, Cdim);
        fb_init<<<(Nnode * Cdim) / 256, 256, 0, stream>>>(Xb, Fb, b);

        // A_l: column softmax of leaky(u_i + v_j + ang_ij), recomputed on the fly
        el_stats<<<Nnode, 256, 0, stream>>>(us, vs, cph, sph, Wgamma, colmax, colsum);
        el_fill<<<dim3(Nnode / 256, Nnode), 256, 0, stream>>>(us, vs, cph, sph, Wgamma,
                                                              colmax, colsum, Abf);
        // l_all = A_l @ Y_l, per-head ReLU-sum -> Fb[:,512:768]
        launch_gemm(Abf, Nnode, Yt + (size_t)768 * Nnode, Nnode, C768, nullptr, 768, 1,
                    Nnode, 768, Nnode, stream);
        relu_heads<<<(Nnode * Cdim) / 256, 256, 0, stream>>>(C768, Fb, 512);

        // out_down = fused @ fuse_w^T ; then +bias, x0.25, 2x2 upsample
        launch_gemm(Fb, 768, Wfuse, 768, down, nullptr, Cdim, 1, Nnode, Cdim, 768, stream);
        upsample_kernel<<<(Cdim * 128 * 128) / 256, 256, 0, stream>>>(down, fuseb, out, b);
    }
}
